// SelectiveAttention_82643760709864
// MI455X (gfx1250) — matrix operand order repaired and checked by reading, not compiled, not run on hardware
//
#include <hip/hip_runtime.h>
#include <hip/hip_bf16.h>

// CDNA5 / gfx1250 fused causal flash-attention with cache prefix.
// H=16 heads, D=64, T=2048 queries, 1024 cache rows, T_full=3072.
// f16 WMMA (v_wmma_f32_16x16x32_f16), f32 accumulation + softmax stats.
// Double-buffered K/V tiles in LDS, deferred row-sum reduction (FA2 style).

typedef __attribute__((ext_vector_type(16))) _Float16 v16h;
typedef __attribute__((ext_vector_type(8)))  _Float16 v8h;
typedef __attribute__((ext_vector_type(8)))  float    v8f;

#define NHEADS     16
#define HEADDIM    64
#define TQ         2048
#define TCACHE     1024
#define EMB        1024
#define ROWSTRIDE  3072      // 3*EMB floats per row of qkv / cache
#define BR         64        // query rows per block
#define BC         64        // key rows per tile
#define NWAVES     4         // 128 threads, wave32

__global__ __launch_bounds__(128) void
fa_cdna5_kernel(const float* __restrict__ qkv,
                const float* __restrict__ cache,
                float* __restrict__ out)
{
    __shared__ __align__(32) _Float16 qs[BR * HEADDIM];             // Q tile (scaled f16)
    __shared__ __align__(32) _Float16 ks[2][BC * HEADDIM];          // K tiles, row-major (ping-pong)
    __shared__ __align__(32) _Float16 vts[2][HEADDIM * BC];         // V tiles, transposed [d][s]
    __shared__ __align__(32) _Float16 ps[NWAVES * 16 * BC];         // per-wave P slab

    const int b     = blockIdx.x;            // query tile index (0..31)
    const int h     = blockIdx.y;            // head (0..15)
    const int tid   = threadIdx.x;           // 0..127
    const int wave  = tid >> 5;              // 0..3
    const int lane  = tid & 31;
    const int lhalf = lane >> 4;             // 0/1 (lane half)
    const int l15   = lane & 15;

    // fold 1/sqrt(64) and log2(e) into Q so softmax is a bare exp2
    const float QSCALE = 0.125f * 1.44269504088896341f;

    const int ktmax  = 17 + b;     // key tiles 0 .. 16+b (last one diagonal)
    const int ktdiag = 16 + b;

    // ---- cooperative K/V tile stage: global f32 -> LDS f16 (V transposed) ----
    auto load_tile = [&](int kt, int buf) {
        const bool   incache = (kt < 16);
        const float* src     = incache ? cache : qkv;
        const int    rowoff  = incache ? kt * BC : kt * BC - TCACHE;
        const float* kp = src + (size_t)rowoff * ROWSTRIDE + EMB + h * HEADDIM;
        const float* vp = kp + EMB;
        #pragma unroll
        for (int i = 0; i < 8; ++i) {
            int idx = tid + i * 128;              // 0..1023
            int r   = idx >> 4;                   // row 0..63
            int c4  = (idx & 15) << 2;            // col 0,4,..,60
            const float4 fk = *(const float4*)(kp + (size_t)r * ROWSTRIDE + c4);
            _Float16* kd = &ks[buf][r * HEADDIM + c4];
            kd[0] = (_Float16)fk.x; kd[1] = (_Float16)fk.y;
            kd[2] = (_Float16)fk.z; kd[3] = (_Float16)fk.w;
            const float4 fv = *(const float4*)(vp + (size_t)r * ROWSTRIDE + c4);
            vts[buf][(c4 + 0) * BC + r] = (_Float16)fv.x;
            vts[buf][(c4 + 1) * BC + r] = (_Float16)fv.y;
            vts[buf][(c4 + 2) * BC + r] = (_Float16)fv.z;
            vts[buf][(c4 + 3) * BC + r] = (_Float16)fv.w;
        }
    };

    // ---------------- stage Q tile -> LDS (f16, pre-scaled) ----------------
    {
        const float* qp = qkv + (size_t)(b * BR) * ROWSTRIDE + h * HEADDIM;
        #pragma unroll
        for (int i = 0; i < 8; ++i) {
            int idx = tid + i * 128;
            int r   = idx >> 4;
            int c4  = (idx & 15) << 2;
            const float4 f = *(const float4*)(qp + (size_t)r * ROWSTRIDE + c4);
            _Float16* d = &qs[r * HEADDIM + c4];
            d[0] = (_Float16)(f.x * QSCALE);
            d[1] = (_Float16)(f.y * QSCALE);
            d[2] = (_Float16)(f.z * QSCALE);
            d[3] = (_Float16)(f.w * QSCALE);
        }
    }
    load_tile(0, 0);               // pipeline prologue: first K/V tile into buffer 0
    __syncthreads();

    // -------- each wave keeps its Q A-operands (16 rows x 64) in VGPRs --------
    // A-layout (16-bit, 16x32): lane L: M=L&15; halves 0..15 = K[hi*16..+15]
    // (hi = L>>4), matching the B operand's k order
    v16h qa[2];
    #pragma unroll
    for (int kc = 0; kc < 2; ++kc) {
        int m  = wave * 16 + l15;
        int o1 = m * HEADDIM + kc * 32 + lhalf * 16;
        v8h lo  = *(const v8h*)&qs[o1];
        v8h hi8 = *(const v8h*)&qs[o1 + 8];
        v16h a;
        #pragma unroll
        for (int j = 0; j < 8; ++j) { a[j] = lo[j]; a[8 + j] = hi8[j]; }
        qa[kc] = a;
    }

    v8f o_acc[4];
    #pragma unroll
    for (int dt = 0; dt < 4; ++dt) {
        v8f z = {};
        o_acc[dt] = z;
    }
    float mstat[8], lpart[8];      // lpart: per-lane PARTIAL row sum (reduced at end)
    #pragma unroll
    for (int r = 0; r < 8; ++r) { mstat[r] = -1e30f; lpart[r] = 0.0f; }

    for (int kt = 0; kt < ktmax; ++kt) {
        const int cur = kt & 1;

        // ---- overlap: stage NEXT tile into the idle buffer while computing ----
        if (kt + 1 < ktmax) load_tile(kt + 1, cur ^ 1);
        if (kt + 2 < ktmax) {      // L2 prefetch two tiles ahead (global_prefetch_b8)
            const bool  nc = (kt + 2) < 16;
            const float* nb = nc ? cache : qkv;
            int nro = nc ? (kt + 2) * BC : (kt + 2) * BC - TCACHE;
            __builtin_prefetch(nb + (size_t)(nro + (tid >> 1)) * ROWSTRIDE
                                  + EMB + (tid & 1) * EMB + h * HEADDIM, 0, 1);
        }

        // ---------------- S = Q * K^T  (4 column tiles of 16 keys) ----------------
        // B operand (32x16, same k order as A): lane L: N=L&15; 16 halves = K[hi*16..+15]
        v8f s[4];
        #pragma unroll
        for (int nt = 0; nt < 4; ++nt) {
            v8f acc = {};
            #pragma unroll
            for (int kc = 0; kc < 2; ++kc) {
                int nrow = nt * 16 + l15;
                const v16h bm =
                    *(const v16h*)&ks[cur][nrow * HEADDIM + kc * 32 + lhalf * 16];
                acc = __builtin_amdgcn_wmma_f32_16x16x32_f16(
                          false, qa[kc], false, bm, (short)0, acc, false, false);
            }
            s[nt] = acc;
        }

        // ---------------- causal mask (diagonal tile only) ----------------
        if (kt == ktdiag) {
            #pragma unroll
            for (int nt = 0; nt < 4; ++nt) {
                #pragma unroll
                for (int r = 0; r < 8; ++r) {
                    int m = wave * 16 + r + 8 * lhalf;   // query row in 64-block
                    int n = nt * 16 + l15;               // key row in 64-block
                    if (n > m) s[nt][r] = -1e30f;
                }
            }
        }

        // -------- online softmax: exact row max (shfl), DEFERRED row sum --------
        float alpha[8];
        #pragma unroll
        for (int r = 0; r < 8; ++r) {
            float mx = fmaxf(fmaxf(s[0][r], s[1][r]), fmaxf(s[2][r], s[3][r]));
            #pragma unroll
            for (int off = 8; off >= 1; off >>= 1)      // within 16-lane half
                mx = fmaxf(mx, __shfl_xor(mx, off, 32));
            float mnew = fmaxf(mstat[r], mx);
            alpha[r]  = exp2f(mstat[r] - mnew);
            mstat[r]  = mnew;
            float rs = 0.0f;
            #pragma unroll
            for (int nt = 0; nt < 4; ++nt) {
                float p = exp2f(s[nt][r] - mnew);
                s[nt][r] = p;
                rs += p;
            }
            lpart[r] = lpart[r] * alpha[r] + rs;        // per-lane partial, no shfl
        }

        // rescale running O
        #pragma unroll
        for (int dt = 0; dt < 4; ++dt)
            #pragma unroll
            for (int r = 0; r < 8; ++r)
                o_acc[dt][r] *= alpha[r];

        // -------- P: C-layout -> A-layout via wave-private LDS slab --------
        _Float16* pw = &ps[wave * 16 * BC];
        #pragma unroll
        for (int nt = 0; nt < 4; ++nt) {
            #pragma unroll
            for (int r = 0; r < 8; ++r) {
                int m    = r + 8 * lhalf;
                int sidx = nt * 16 + l15;
                pw[m * BC + sidx] = (_Float16)s[nt][r];
            }
        }
        // LDS ops are in-order within a wave; slab is wave-private, no barrier.

        // ---------------- O += P * V ----------------
        #pragma unroll
        for (int sc = 0; sc < 2; ++sc) {
            int o1 = l15 * BC + sc * 32 + lhalf * 16;
            v8h lo  = *(const v8h*)&pw[o1];
            v8h hi8 = *(const v8h*)&pw[o1 + 8];
            v16h pa;
            #pragma unroll
            for (int j = 0; j < 8; ++j) { pa[j] = lo[j]; pa[8 + j] = hi8[j]; }
            #pragma unroll
            for (int dt = 0; dt < 4; ++dt) {
                const v16h bv =
                    *(const v16h*)&vts[cur][(dt * 16 + l15) * BC + sc * 32 + lhalf * 16];
                o_acc[dt] = __builtin_amdgcn_wmma_f32_16x16x32_f16(
                                false, pa, false, bv, (short)0, o_acc[dt], false, false);
            }
        }

        __syncthreads();   // next-tile stores visible; current buffer free for kt+2
    }

    // -------- epilogue: reduce deferred row sums once, normalize, store --------
    float linv[8];
    #pragma unroll
    for (int r = 0; r < 8; ++r) {
        float l = lpart[r];
        #pragma unroll
        for (int off = 8; off >= 1; off >>= 1)
            l += __shfl_xor(l, off, 32);
        linv[r] = 1.0f / l;
    }

    const int rowbase = b * BR + wave * 16;
    #pragma unroll
    for (int dt = 0; dt < 4; ++dt) {
        #pragma unroll
        for (int r = 0; r < 8; ++r) {
            int m   = r + 8 * lhalf;
            int row = rowbase + m;
            int col = h * HEADDIM + dt * 16 + l15;
            __builtin_nontemporal_store(o_acc[dt][r] * linv[r],
                                        &out[(size_t)row * EMB + col]);
        }
    }
}

extern "C" void kernel_launch(void* const* d_in, const int* in_sizes, int n_in,
                              void* d_out, int out_size, void* d_ws, size_t ws_size,
                              hipStream_t stream) {
    const float* qkv   = (const float*)d_in[0];   // [2048, 3072] fp32
    const float* cache = (const float*)d_in[1];   // [1024, 3072] fp32
    float* out = (float*)d_out;                   // [2048, 1024] fp32

    dim3 grid(TQ / BR, NHEADS);   // (32, 16)
    dim3 block(128);              // 4 waves (wave32)
    fa_cdna5_kernel<<<grid, block, 0, stream>>>(qkv, cache, out);
}
